// QuantTransEncLDARTSCell_26611617366526
// MI455X (gfx1250) — matrix operand order repaired and checked by reading, not compiled, not run on hardware
//
#include <hip/hip_runtime.h>
#include <hip/hip_bf16.h>
#include <math.h>

#define S_ 1024
#define B_ 8
#define D_ 1024
#define H_ 16
#define HD_ 64
#define F_ 4096
#define D3_ 3072
#define M_ (S_*B_)

typedef __bf16 bf16;
typedef __bf16 v16bf __attribute__((ext_vector_type(16)));
typedef __bf16 v8bf  __attribute__((ext_vector_type(8)));
typedef float  v8f   __attribute__((ext_vector_type(8)));
typedef int    v4i   __attribute__((ext_vector_type(4)));

// ---------- CDNA5 async global->LDS copy (ASYNCcnt-tracked) ----------

#if __has_builtin(__builtin_amdgcn_global_load_async_to_lds_b128)
#define HAVE_ASYNC_LDS 1
#else
#define HAVE_ASYNC_LDS 0
#endif

typedef __attribute__((address_space(1))) v4i gv4i;   // global int4
typedef __attribute__((address_space(3))) v4i lv4i;   // LDS int4

__device__ __forceinline__ void lds_copy_b128(const bf16* g, bf16* l) {
#if HAVE_ASYNC_LDS
  __builtin_amdgcn_global_load_async_to_lds_b128((gv4i*)(void*)g, (lv4i*)l, 0, 0);
#else
  *(v8bf*)l = *(const v8bf*)g;
#endif
}

// wait for outstanding async LDS copies (if any), then workgroup barrier
__device__ __forceinline__ void async_join() {
#if HAVE_ASYNC_LDS
#if __has_builtin(__builtin_amdgcn_s_wait_asynccnt)
  __builtin_amdgcn_s_wait_asynccnt(0);
#else
  asm volatile("s_wait_asynccnt 0x0" ::: "memory");
#endif
#endif
  __syncthreads();
}

// ---------- WMMA helpers (CDNA5 wave32, 16x16x32 bf16) ----------

__device__ __forceinline__ v8f wmma_bf16(v16bf a, v16bf b, v8f c) {
  return __builtin_amdgcn_wmma_f32_16x16x32_bf16(
      /*neg_a=*/false, a, /*neg_b=*/false, b,
      /*c_mod=*/(short)0, c, /*reuse_a=*/false, /*reuse_b=*/false);
}

// A-matrix 16x32 bf16 fragment (ISA 7.12.2): lane half 0 covers K 0..7 and
// 16..23; half 1 covers K 8..15 and 24..31.
__device__ __forceinline__ v16bf frag_a(const bf16* rowp, int half) {
  const bf16* p = rowp + half * 8;
  v8bf lo = *(const v8bf*)(p);
  v8bf hi = *(const v8bf*)(p + 16);
  return __builtin_shufflevector(lo, hi, 0,1,2,3,4,5,6,7,8,9,10,11,12,13,14,15);
}

// B-matrix 32x16 bf16 fragment: lane half 0 covers K 0..7 and 16..23; half 1 covers K 8..15 and 24..31.
__device__ __forceinline__ v16bf frag_b(const bf16* rowp, int half) {
  const bf16* p = rowp + half * 8;
  v8bf lo = *(const v8bf*)(p);
  v8bf hi = *(const v8bf*)(p + 16);
  return __builtin_shufflevector(lo, hi, 0,1,2,3,4,5,6,7,8,9,10,11,12,13,14,15);
}

// ---------- small prep kernels ----------

__global__ void softmax_alpha_k(const float* __restrict__ alpha,
                                float* __restrict__ probs) {
  if (threadIdx.x == 0) {
    for (int r = 0; r < 3; ++r) {
      float m = -1e30f;
      for (int i = 0; i < 4; ++i) m = fmaxf(m, alpha[r*4+i]);
      float e[4], s = 0.f;
      for (int i = 0; i < 4; ++i) { e[i] = __expf(alpha[r*4+i] - m); s += e[i]; }
      for (int i = 0; i < 4; ++i) probs[r*4+i] = e[i] / s;
    }
  }
}

__global__ __launch_bounds__(256)
void absmax_sum_part(const float* __restrict__ w, long long n,
                     float* __restrict__ pmax, float* __restrict__ psum) {
  __shared__ float smax[256], ssum[256];
  float mx = 0.f, sm = 0.f;
  for (long long i = (long long)blockIdx.x*256 + threadIdx.x; i < n;
       i += (long long)gridDim.x*256) {
    float a = fabsf(w[i]); mx = fmaxf(mx, a); sm += a;
  }
  smax[threadIdx.x] = mx; ssum[threadIdx.x] = sm; __syncthreads();
  for (int s = 128; s > 0; s >>= 1) {
    if (threadIdx.x < s) {
      smax[threadIdx.x] = fmaxf(smax[threadIdx.x], smax[threadIdx.x+s]);
      ssum[threadIdx.x] += ssum[threadIdx.x+s];
    }
    __syncthreads();
  }
  if (threadIdx.x == 0) { pmax[blockIdx.x] = smax[0]; psum[blockIdx.x] = ssum[0]; }
}

__global__ __launch_bounds__(256)
void stats_finish(const float* __restrict__ pmax, const float* __restrict__ psum,
                  int nb, float invn, float* __restrict__ out2) {
  __shared__ float smax[256], ssum[256];
  float mx = 0.f, sm = 0.f;
  for (int i = threadIdx.x; i < nb; i += 256) { mx = fmaxf(mx, pmax[i]); sm += psum[i]; }
  smax[threadIdx.x] = mx; ssum[threadIdx.x] = sm; __syncthreads();
  for (int s = 128; s > 0; s >>= 1) {
    if (threadIdx.x < s) {
      smax[threadIdx.x] = fmaxf(smax[threadIdx.x], smax[threadIdx.x+s]);
      ssum[threadIdx.x] += ssum[threadIdx.x+s];
    }
    __syncthreads();
  }
  if (threadIdx.x == 0) { out2[0] = smax[0]; out2[1] = ssum[0] * invn; }
}

__device__ __forceinline__ float quant_val(float w, float amax, float amean, int nbit) {
  if (nbit == 1) return (w > 0.f) ? amean : ((w < 0.f) ? -amean : 0.f);
  float qmax = (float)((1 << (nbit - 1)) - 1);
  float scale = amax / qmax;
  if (scale < 1e-30f) return 0.f;
  float q = rintf(w / scale);
  q = fminf(fmaxf(q, -qmax - 1.f), qmax);
  return q * scale;
}

__global__ __launch_bounds__(256)
void quant_branch(const float* __restrict__ w, bf16* __restrict__ out,
                  long long n, const float* __restrict__ st, int nbit) {
  float amax = st[0], amean = st[1];
  for (long long i = (long long)blockIdx.x*256 + threadIdx.x; i < n;
       i += (long long)gridDim.x*256)
    out[i] = (bf16)quant_val(w[i], amax, amean, nbit);
}

__global__ __launch_bounds__(256)
void quant_mix(const float* __restrict__ w, const float* __restrict__ st8,
               const float* __restrict__ p4, bf16* __restrict__ out, long long n) {
  float p0 = p4[0], p1 = p4[1], p2 = p4[2], p3 = p4[3];
  float a0 = st8[0], m0 = st8[1], a1 = st8[2], m1 = st8[3];
  float a2 = st8[4], m2 = st8[5], a3 = st8[6], m3 = st8[7];
  for (long long j = (long long)blockIdx.x*256 + threadIdx.x; j < n;
       j += (long long)gridDim.x*256) {
    float acc = p0 * quant_val(w[j],       a0, m0, 1);
    acc      += p1 * quant_val(w[n + j],   a1, m1, 2);
    acc      += p2 * quant_val(w[2*n + j], a2, m2, 4);
    acc      += p3 * quant_val(w[3*n + j], a3, m3, 8);
    out[j] = (bf16)acc;
  }
}

__global__ void mix_bias(const float* __restrict__ b, const float* __restrict__ p4,
                         float* __restrict__ out, int n) {
  int i = blockIdx.x*256 + threadIdx.x;
  if (i < n) out[i] = p4[0]*b[i] + p4[1]*b[n+i] + p4[2]*b[2*n+i] + p4[3]*b[3*n+i];
}

__global__ __launch_bounds__(256)
void f2bf(const float* __restrict__ in, bf16* __restrict__ out, long long n) {
  for (long long i = (long long)blockIdx.x*256 + threadIdx.x; i < n;
       i += (long long)gridDim.x*256)
    out[i] = (bf16)in[i];
}

// ---------- bf16 TN GEMM:  C[M,N] = A[M,K] * Bw[N,K]^T (+bias) ----------
// Block tile 128x128x32, 8 waves, wave tile 32x64 (2x4 WMMA accumulators),
// double-buffered LDS fed by async global->LDS copies.
// MODE 0: store (C+bias) as bf16        (qkv projection)
// MODE 1: outF += scale_p[0]*(C+bias)   (out-proj accumulation into src2)
// MODE 2: store gelu(C+bias) as bf16    (FFN linear1)
// MODE 3: store (C+bias) as f32         (FFN linear2)
template<int MODE>
__global__ __launch_bounds__(256)
void gemm_tn(const bf16* __restrict__ A, const bf16* __restrict__ Bw,
             const float* __restrict__ bias, float* __restrict__ outF,
             bf16* __restrict__ outBf, const float* __restrict__ scale_p,
             int M, int N, int K) {
  constexpr int LDT = 40;                 // padded stride, 16B-aligned rows
  __shared__ bf16 As[2][128 * LDT];
  __shared__ bf16 Bs[2][128 * LDT];
  const int t = threadIdx.x, lane = t & 31, wave = t >> 5;
  const int wr = wave & 3, wc = wave >> 2;
  const int half = lane >> 4, l15 = lane & 15;
  const int bm = blockIdx.y * 128, bn = blockIdx.x * 128;
  const int lrow = t >> 2, lseg = t & 3;  // 64 rows x 4 segs per 256 threads
  v8f acc[2][4] = {};

  auto load_tile = [&](int k0, int buf) {
#pragma unroll
    for (int i = 0; i < 2; ++i) {
      int row = lrow + i * 64;
      lds_copy_b128(A  + (long long)(bm + row) * K + k0 + lseg * 8,
                    &As[buf][row * LDT + lseg * 8]);
      lds_copy_b128(Bw + (long long)(bn + row) * K + k0 + lseg * 8,
                    &Bs[buf][row * LDT + lseg * 8]);
    }
  };

  load_tile(0, 0);
  async_join();

  const int nk = K >> 5;
  for (int kt = 0; kt < nk; ++kt) {
    if (kt + 1 < nk) load_tile((kt + 1) << 5, (kt + 1) & 1);
    const bf16* Ab = As[kt & 1];
    const bf16* Bb = Bs[kt & 1];
    v16bf af[2], bfv[4];
#pragma unroll
    for (int i = 0; i < 2; ++i)
      af[i] = frag_a(Ab + (wr * 32 + i * 16 + l15) * LDT, half);
#pragma unroll
    for (int j = 0; j < 4; ++j)
      bfv[j] = frag_b(Bb + (wc * 64 + j * 16 + l15) * LDT, half);
#pragma unroll
    for (int i = 0; i < 2; ++i)
#pragma unroll
      for (int j = 0; j < 4; ++j)
        acc[i][j] = wmma_bf16(af[i], bfv[j], acc[i][j]);
    async_join();                         // next tile resident + LDS reusable
  }

  float scl = 1.0f;
  if constexpr (MODE == 1) scl = scale_p[0];
#pragma unroll
  for (int i = 0; i < 2; ++i)
#pragma unroll
    for (int j = 0; j < 4; ++j)
#pragma unroll
      for (int r = 0; r < 8; ++r) {
        int m = bm + wr * 32 + i * 16 + half * 8 + r;
        int n = bn + wc * 64 + j * 16 + l15;
        float c = acc[i][j][r] + (bias ? bias[n] : 0.0f);
        long long o = (long long)m * N + n;
        if constexpr (MODE == 0) {
          outBf[o] = (bf16)c;
        } else if constexpr (MODE == 1) {
          outF[o] += scl * c;
        } else if constexpr (MODE == 2) {
          float g = 0.5f * c * (1.0f + erff(c * 0.70710678118654752f));
          outBf[o] = (bf16)g;
        } else {
          outF[o] = c;
        }
      }
}

// ---------- flash attention (one branch) ----------
// qkv: bf16 [S*B, 3D], q at col h*64, k at D+h*64, v at 2D+h*64.
// grid = (S/128, B*H), 256 threads = 8 waves, each wave owns 16 query rows.
__global__ __launch_bounds__(256)
void flash_attn(const bf16* __restrict__ qkv, bf16* __restrict__ o) {
  constexpr int LQ = 72;                  // padded row stride (bf16 elems)
  __shared__ bf16 Qs[128 * LQ];
  __shared__ bf16 Ks[64 * LQ];
  __shared__ bf16 Vt[64 * LQ];            // V transposed: [d][t]
  __shared__ bf16 Ps[128 * LQ];           // per-wave 16-row P tiles
  const int t = threadIdx.x, lane = t & 31, wave = t >> 5;
  const int half = lane >> 4, l15 = lane & 15;
  const int bh = blockIdx.y;
  const int b = bh / H_, h = bh % H_;
  const int s0 = blockIdx.x * 128;
  const int qcol = h * HD_;

#pragma unroll
  for (int i = 0; i < 4; ++i) {           // async load 128x64 Q block
    int idx = t + i * 256, row = idx >> 3, seg = idx & 7;
    long long src = ((long long)(s0 + row) * B_ + b) * D3_ + qcol + seg * 8;
    lds_copy_b128(qkv + src, Qs + row * LQ + seg * 8);
  }

  v8f oacc[4] = {};
  float mprev[8], lsum[8];
#pragma unroll
  for (int r = 0; r < 8; ++r) { mprev[r] = -1e30f; lsum[r] = 0.f; }
  bf16* Pw = Ps + wave * 16 * LQ;

  for (int kt = 0; kt < S_; kt += 64) {
    __syncthreads();                      // previous PV reads of Ks/Vt done
#pragma unroll
    for (int i = 0; i < 2; ++i) {         // load 64x64 K (async) and V^T
      int idx = t + i * 256, row = idx >> 3, seg = idx & 7;
      long long ks = ((long long)(kt + row) * B_ + b) * D3_ + D_ + qcol + seg * 8;
      lds_copy_b128(qkv + ks, Ks + row * LQ + seg * 8);
      long long vs = ((long long)(kt + row) * B_ + b) * D3_ + 2 * D_ + qcol + seg * 8;
      v8bf vv = *(const v8bf*)(qkv + vs);
#pragma unroll
      for (int j = 0; j < 8; ++j) Vt[(seg * 8 + j) * LQ + row] = vv[j];
    }
    async_join();                         // Q (first iter) + K resident

    // scores S = Q K^T : per-wave 16x64 in 4 accumulators
    v8f sacc[4] = {};
#pragma unroll
    for (int ks = 0; ks < 2; ++ks) {
      v16bf aq = frag_a(Qs + (wave * 16 + l15) * LQ + ks * 32, half);
#pragma unroll
      for (int nt = 0; nt < 4; ++nt) {
        v16bf bk = frag_b(Ks + (nt * 16 + l15) * LQ + ks * 32, half);
        sacc[nt] = wmma_bf16(aq, bk, sacc[nt]);
      }
    }

    const float scale = 0.125f;           // 1/sqrt(64)
#pragma unroll
    for (int r = 0; r < 8; ++r) {
      float mx = -1e30f;
#pragma unroll
      for (int nt = 0; nt < 4; ++nt) mx = fmaxf(mx, sacc[nt][r]);
      mx *= scale;
#pragma unroll
      for (int msk = 1; msk < 16; msk <<= 1)
        mx = fmaxf(mx, __shfl_xor(mx, msk, 32));
      float mnew = fmaxf(mprev[r], mx);
      float corr = __expf(mprev[r] - mnew);
      float rs = 0.f;
#pragma unroll
      for (int nt = 0; nt < 4; ++nt) {
        float p = __expf(sacc[nt][r] * scale - mnew);
        Pw[(half * 8 + r) * LQ + nt * 16 + l15] = (bf16)p;
        rs += p;
      }
#pragma unroll
      for (int msk = 1; msk < 16; msk <<= 1) rs += __shfl_xor(rs, msk, 32);
      lsum[r] = lsum[r] * corr + rs;
      mprev[r] = mnew;
#pragma unroll
      for (int dt = 0; dt < 4; ++dt) oacc[dt][r] *= corr;
    }

    // O += P V : P is wave-private in LDS (A layout), Vt gives B layout
#pragma unroll
    for (int ks = 0; ks < 2; ++ks) {
      v16bf ap = frag_a(Pw + l15 * LQ + ks * 32, half);
#pragma unroll
      for (int dt = 0; dt < 4; ++dt) {
        v16bf bv = frag_b(Vt + (dt * 16 + l15) * LQ + ks * 32, half);
        oacc[dt] = wmma_bf16(ap, bv, oacc[dt]);
      }
    }
  }

#pragma unroll
  for (int dt = 0; dt < 4; ++dt)
#pragma unroll
    for (int r = 0; r < 8; ++r) {
      int s = s0 + wave * 16 + half * 8 + r;
      int d = dt * 16 + l15;
      long long dst = ((long long)s * B_ + b) * D_ + qcol + d;
      o[dst] = (bf16)(oacc[dt][r] / lsum[r]);
    }
}

// ---------- residual add + layernorm ----------
__global__ __launch_bounds__(256)
void layernorm_add(const float* __restrict__ a, const float* __restrict__ res,
                   const float* __restrict__ g, const float* __restrict__ be,
                   float* __restrict__ outF, bf16* __restrict__ outBf) {
  __shared__ float red[256];
  const int row = blockIdx.x, t = threadIdx.x;
  const long long base = (long long)row * D_;
  float x[4]; float sm = 0.f;
#pragma unroll
  for (int i = 0; i < 4; ++i) {
    x[i] = a[base + t + i * 256] + res[base + t + i * 256];
    sm += x[i];
  }
  red[t] = sm; __syncthreads();
  for (int s = 128; s > 0; s >>= 1) { if (t < s) red[t] += red[t + s]; __syncthreads(); }
  float mean = red[0] * (1.0f / D_); __syncthreads();
  float vs = 0.f;
#pragma unroll
  for (int i = 0; i < 4; ++i) { float d = x[i] - mean; vs += d * d; }
  red[t] = vs; __syncthreads();
  for (int s = 128; s > 0; s >>= 1) { if (t < s) red[t] += red[t + s]; __syncthreads(); }
  float inv = rsqrtf(red[0] * (1.0f / D_) + 1e-5f);
#pragma unroll
  for (int i = 0; i < 4; ++i) {
    int c = t + i * 256;
    float v = (x[i] - mean) * inv * g[c] + be[c];
    if (outF)  outF[base + c] = v;
    if (outBf) outBf[base + c] = (bf16)v;
  }
}

// ---------- host launch ----------
extern "C" void kernel_launch(void* const* d_in, const int* in_sizes, int n_in,
                              void* d_out, int out_size, void* d_ws, size_t ws_size,
                              hipStream_t stream) {
  (void)in_sizes; (void)n_in; (void)out_size; (void)ws_size;
  const float* src   = (const float*)d_in[0];
  const float* alpha = (const float*)d_in[1];
  const float* inW   = (const float*)d_in[2];
  const float* inB   = (const float*)d_in[3];
  const float* opW   = (const float*)d_in[4];
  const float* opB   = (const float*)d_in[5];
  const float* w1    = (const float*)d_in[6];
  const float* b1    = (const float*)d_in[7];
  const float* w2    = (const float*)d_in[8];
  const float* b2    = (const float*)d_in[9];
  const float* g1    = (const float*)d_in[10];
  const float* be1   = (const float*)d_in[11];
  const float* g2    = (const float*)d_in[12];
  const float* be2   = (const float*)d_in[13];
  float* out = (float*)d_out;

  char* wptr = (char*)d_ws;
  auto alloc = [&](size_t bytes) -> char* {
    char* p = wptr; wptr += (bytes + 255) & ~(size_t)255; return p;
  };
  float* probs = (float*)alloc(12 * 4);
  float* stats = (float*)alloc(32 * 4);
  float* pmax  = (float*)alloc(512 * 4);
  float* psum  = (float*)alloc(512 * 4);
  float* b1e   = (float*)alloc((size_t)F_ * 4);
  float* b2e   = (float*)alloc((size_t)D_ * 4);
  bf16* winq   = (bf16*)alloc((size_t)4 * D3_ * D_ * 2);
  bf16* wopq   = (bf16*)alloc((size_t)4 * D_ * D_ * 2);
  bf16* w1e    = (bf16*)alloc((size_t)F_ * D_ * 2);
  bf16* w2e    = (bf16*)alloc((size_t)D_ * F_ * 2);
  bf16* srcb   = (bf16*)alloc((size_t)M_ * D_ * 2);
  bf16* qkvb   = (bf16*)alloc((size_t)M_ * D3_ * 2);
  bf16* ob     = (bf16*)alloc((size_t)M_ * D_ * 2);
  float* src2  = (float*)alloc((size_t)M_ * D_ * 4);
  float* x     = (float*)alloc((size_t)M_ * D_ * 4);
  bf16* xb     = (bf16*)alloc((size_t)M_ * D_ * 2);
  bf16* hb     = (bf16*)alloc((size_t)M_ * F_ * 2);
  float* y = src2;  // src2 dead after LN1; reuse for FFN output

  softmax_alpha_k<<<1, 32, 0, stream>>>(alpha, probs);

  const long long ns[4] = {(long long)D3_ * D_, (long long)D_ * D_,
                           (long long)F_ * D_, (long long)D_ * F_};
  const float* bases[4] = {inW, opW, w1, w2};
  int combo = 0;
  for (int ts = 0; ts < 4; ++ts)
    for (int i = 0; i < 4; ++i, ++combo) {
      const float* p = bases[ts] + (long long)i * ns[ts];
      absmax_sum_part<<<512, 256, 0, stream>>>(p, ns[ts], pmax, psum);
      stats_finish<<<1, 256, 0, stream>>>(pmax, psum, 512,
                                          1.0f / (float)ns[ts], stats + combo * 2);
    }

  static const int NB[4] = {1, 2, 4, 8};
  for (int i = 0; i < 4; ++i)
    quant_branch<<<1024, 256, 0, stream>>>(inW + (long long)i * D3_ * D_,
                                           winq + (long long)i * D3_ * D_,
                                           (long long)D3_ * D_, stats + i * 2, NB[i]);
  for (int i = 0; i < 4; ++i)
    quant_branch<<<1024, 256, 0, stream>>>(opW + (long long)i * D_ * D_,
                                           wopq + (long long)i * D_ * D_,
                                           (long long)D_ * D_, stats + (4 + i) * 2, NB[i]);
  quant_mix<<<1024, 256, 0, stream>>>(w1, stats + 16, probs + 4, w1e, (long long)F_ * D_);
  quant_mix<<<1024, 256, 0, stream>>>(w2, stats + 24, probs + 8, w2e, (long long)D_ * F_);
  mix_bias<<<(F_ + 255) / 256, 256, 0, stream>>>(b1, probs + 4, b1e, F_);
  mix_bias<<<(D_ + 255) / 256, 256, 0, stream>>>(b2, probs + 8, b2e, D_);
  f2bf<<<2048, 256, 0, stream>>>(src, srcb, (long long)M_ * D_);
  (void)hipMemsetAsync(src2, 0, (size_t)M_ * D_ * 4, stream);

  for (int i = 0; i < 4; ++i) {
    gemm_tn<0><<<dim3(D3_ / 128, M_ / 128), 256, 0, stream>>>(
        srcb, winq + (long long)i * D3_ * D_, inB + (long long)i * D3_,
        nullptr, qkvb, nullptr, M_, D3_, D_);
    flash_attn<<<dim3(S_ / 128, B_ * H_), 256, 0, stream>>>(qkvb, ob);
    gemm_tn<1><<<dim3(D_ / 128, M_ / 128), 256, 0, stream>>>(
        ob, wopq + (long long)i * D_ * D_, opB + (long long)i * D_,
        src2, nullptr, probs + i, M_, D_, D_);
  }

  layernorm_add<<<M_, 256, 0, stream>>>(src, src2, g1, be1, x, xb);
  gemm_tn<2><<<dim3(F_ / 128, M_ / 128), 256, 0, stream>>>(
      xb, w1e, b1e, nullptr, hb, nullptr, M_, F_, D_);
  gemm_tn<3><<<dim3(D_ / 128, M_ / 128), 256, 0, stream>>>(
      hb, w2e, b2e, y, nullptr, nullptr, M_, D_, F_);
  layernorm_add<<<M_, 256, 0, stream>>>(x, y, g2, be2, out, nullptr);
}
